// SE3TransformerLayer_37177236914840
// MI455X (gfx1250) — matrix operand order repaired and checked by reading, not compiled, not run on hardware
//
#include <hip/hip_runtime.h>
#include <hip/hip_bf16.h>
#include <math.h>

typedef __bf16 bf16_t;
typedef __attribute__((ext_vector_type(16))) __bf16 v16bf;
typedef __attribute__((ext_vector_type(8)))  __bf16 v8bf;
typedef __attribute__((ext_vector_type(8)))  float  v8f;

#define LDS_STRIDE 72        // 64-col tile + 8 pad: rows 36 banks apart -> conflict-free b128
#define A_BUF_BYTES (64  * LDS_STRIDE * 2)   // 9216
#define B_BUF_BYTES (128 * LDS_STRIDE * 2)   // 18432

__device__ __forceinline__ v8f wmma_bf16f32(v16bf a, v16bf b, v8f c) {
  return __builtin_amdgcn_wmma_f32_16x16x32_bf16(false, a, false, b, (short)0, c,
                                                 false, false);
}

// CDNA5 async DMA: 32 lanes x 16B straight into LDS, tracked by ASYNCcnt.
__device__ __forceinline__ void async_b128(unsigned int lds_addr, const bf16_t* gaddr) {
  asm volatile("global_load_async_to_lds_b128 %0, %1, off"
               :: "v"(lds_addr), "v"(gaddr) : "memory");
}

enum { EPI_BIAS = 0, EPI_RELU = 1, EPI_ATTN = 2 };

// C[M,N] = A[M,K](bf16 row-major) * B[N,K](bf16 pre-transposed) + epilogue.
// 256 threads (8 waves); block tile 64(M) x 128(N); wave tile 32x32 (2x2 WMMA).
// K staged 64-deep through double-buffered LDS fed by async-to-LDS DMA.
template <int EPI>
__global__ __launch_bounds__(256) void gemm_bf16_wmma(
    const bf16_t* __restrict__ A, long long astride, int lda,
    const bf16_t* __restrict__ B, long long bstride, int ldb,
    const float*  __restrict__ bias,
    float* __restrict__ outF, bf16_t* __restrict__ outB,
    long long cstride, int ldc, int K,
    const float* __restrict__ geo, long long gstride, float scale)
{
  __shared__ alignas(16) bf16_t Asb[2][64  * LDS_STRIDE];
  __shared__ alignas(16) bf16_t Bsb[2][128 * LDS_STRIDE];
  __shared__ float Gr[64 * 3];
  __shared__ float Gc[128 * 3];

  const int tid  = threadIdx.x;
  const int lane = tid & 31;
  const int wid  = tid >> 5;
  const int m0   = blockIdx.y * 64;
  const int n0   = blockIdx.x * 128;
  const int z    = blockIdx.z;

  A += (long long)z * astride;
  B += (long long)z * bstride;

  if (EPI == EPI_ATTN) {
    const float* g = geo + (long long)z * gstride;
    if (tid < 64) {
      Gr[tid * 3 + 0] = g[(m0 + tid) * 3 + 0];
      Gr[tid * 3 + 1] = g[(m0 + tid) * 3 + 1];
      Gr[tid * 3 + 2] = g[(m0 + tid) * 3 + 2];
    }
    if (tid < 128) {
      Gc[tid * 3 + 0] = g[(n0 + tid) * 3 + 0];
      Gc[tid * 3 + 1] = g[(n0 + tid) * 3 + 1];
      Gc[tid * 3 + 2] = g[(n0 + tid) * 3 + 2];
    }
  }

  const int mw = (wid >> 2) * 32;  // wave M offset: 0/32
  const int nw = (wid & 3) * 32;   // wave N offset: 0/32/64/96

  const v8f zero = {0.f, 0.f, 0.f, 0.f, 0.f, 0.f, 0.f, 0.f};
  v8f acc[2][2] = {{zero, zero}, {zero, zero}};

  // async-copy assignments: A 64x64 -> 16 bf16/thread, B 128x64 -> 32 bf16/thread
  const int arow = tid >> 2, ac = (tid & 3) * 16;
  const int brow = tid >> 1, bc = (tid & 1) * 32;
  const unsigned int aBase =
      (unsigned int)(uintptr_t)(&Asb[0][0]) + (unsigned int)(arow * LDS_STRIDE + ac) * 2u;
  const unsigned int bBase =
      (unsigned int)(uintptr_t)(&Bsb[0][0]) + (unsigned int)(brow * LDS_STRIDE + bc) * 2u;

  // per-lane fragment geometry (ISA 7.12.2, wave32)
  const int fm   = lane & 15;
  const int aklo = (lane < 16) ? 0 : 16;  // A: K {0..15} / {16..31}
  const int bklo = (lane < 16) ? 0 : 16;  // B: K {0..15} / {16..31}

  auto issue = [&](int s) {
    const int buf = s & 1;
    const bf16_t* ag = A + (long long)(m0 + arow) * lda + s * 64 + ac;
    const unsigned int al = aBase + (unsigned int)buf * A_BUF_BYTES;
    async_b128(al,      ag);
    async_b128(al + 16, ag + 8);
    const bf16_t* bg = B + (long long)(n0 + brow) * ldb + s * 64 + bc;
    const unsigned int bl = bBase + (unsigned int)buf * B_BUF_BYTES;
    async_b128(bl,      bg);
    async_b128(bl + 16, bg + 8);
    async_b128(bl + 32, bg + 16);
    async_b128(bl + 48, bg + 24);
  };

  auto compute = [&](int buf) {
    const bf16_t* As = Asb[buf];
    const bf16_t* Bs = Bsb[buf];
#pragma unroll
    for (int kk = 0; kk < 64; kk += 32) {
      union { v16bf v; v8bf h[2]; } ua0, ua1, ub0, ub1;
      const bf16_t* pa0 = As + (mw + fm) * LDS_STRIDE + kk + aklo;
      ua0.h[0] = *(const v8bf*)(pa0);
      ua0.h[1] = *(const v8bf*)(pa0 + 8);
      const bf16_t* pa1 = As + (mw + 16 + fm) * LDS_STRIDE + kk + aklo;
      ua1.h[0] = *(const v8bf*)(pa1);
      ua1.h[1] = *(const v8bf*)(pa1 + 8);
      const bf16_t* pb0 = Bs + (nw + fm) * LDS_STRIDE + kk + bklo;
      ub0.h[0] = *(const v8bf*)(pb0);
      ub0.h[1] = *(const v8bf*)(pb0 + 8);
      const bf16_t* pb1 = Bs + (nw + 16 + fm) * LDS_STRIDE + kk + bklo;
      ub1.h[0] = *(const v8bf*)(pb1);
      ub1.h[1] = *(const v8bf*)(pb1 + 8);

      acc[0][0] = wmma_bf16f32(ua0.v, ub0.v, acc[0][0]);
      acc[0][1] = wmma_bf16f32(ua0.v, ub1.v, acc[0][1]);
      acc[1][0] = wmma_bf16f32(ua1.v, ub0.v, acc[1][0]);
      acc[1][1] = wmma_bf16f32(ua1.v, ub1.v, acc[1][1]);
    }
  };

  const int ns = K >> 6;  // K % 64 == 0 for all call sites
  issue(0);
  for (int s = 0; s < ns - 1; ++s) {
    issue(s + 1);                                        // 6 more in flight (12 total)
    asm volatile("s_wait_asynccnt 0x6" ::: "memory");    // stage s landed (in-order)
    __syncthreads();                                     // ...for every wave
    compute(s & 1);
    __syncthreads();                                     // done reading buf before reuse
  }
  asm volatile("s_wait_asynccnt 0x0" ::: "memory");
  __syncthreads();
  compute((ns - 1) & 1);

#pragma unroll
  for (int mi = 0; mi < 2; ++mi) {
#pragma unroll
    for (int ni = 0; ni < 2; ++ni) {
      v8f c = acc[mi][ni];
      const int col = n0 + nw + ni * 16 + (lane & 15);
      float bv = 0.f;
      if (EPI != EPI_ATTN && bias) bv = bias[col];
#pragma unroll
      for (int r = 0; r < 8; ++r) {
        const int row = m0 + mw + mi * 16 + r + ((lane < 16) ? 0 : 8);
        float v = c[r];
        if (EPI == EPI_ATTN) {
          const int li = row - m0, lj = col - n0;
          const float dx = Gr[li * 3 + 0] - Gc[lj * 3 + 0];
          const float dy = Gr[li * 3 + 1] - Gc[lj * 3 + 1];
          const float dz = Gr[li * 3 + 2] - Gc[lj * 3 + 2];
          const float d2 = dx * dx + dy * dy + dz * dz;
          const float dist = (d2 > 0.f) ? sqrtf(d2) : 0.f;
          v = v * scale * __expf(-dist);
        } else {
          v += bv;
          if (EPI == EPI_RELU) v = fmaxf(v, 0.f);
        }
        const long long o = (long long)z * cstride + (long long)row * ldc + col;
        if (outF) outF[o] = v;
        if (outB) outB[o] = (bf16_t)v;
      }
    }
  }
}

// ---------------- row reductions ----------------
__device__ __forceinline__ float block_max(float v, float* red) {
#pragma unroll
  for (int off = 16; off; off >>= 1) v = fmaxf(v, __shfl_xor(v, off, 32));
  if ((threadIdx.x & 31) == 0) red[threadIdx.x >> 5] = v;
  __syncthreads();
  float r = red[0];
#pragma unroll
  for (int i = 1; i < 8; ++i) r = fmaxf(r, red[i]);
  __syncthreads();
  return r;
}
__device__ __forceinline__ float block_sum(float v, float* red) {
#pragma unroll
  for (int off = 16; off; off >>= 1) v += __shfl_xor(v, off, 32);
  if ((threadIdx.x & 31) == 0) red[threadIdx.x >> 5] = v;
  __syncthreads();
  float r = 0.f;
#pragma unroll
  for (int i = 0; i < 8; ++i) r += red[i];
  __syncthreads();
  return r;
}

__global__ __launch_bounds__(256) void softmax_rows(const float* __restrict__ S,
                                                    bf16_t* __restrict__ P, int n) {
  __shared__ float red[8];
  const float* row = S + (long long)blockIdx.x * n;
  bf16_t* prow = P + (long long)blockIdx.x * n;
  float lmax = -INFINITY;
  for (int i = threadIdx.x; i < n; i += 256) lmax = fmaxf(lmax, row[i]);
  const float m = block_max(lmax, red);
  float lsum = 0.f;
  for (int i = threadIdx.x; i < n; i += 256) lsum += __expf(row[i] - m);
  const float s = block_sum(lsum, red);
  const float inv = 1.f / s;
  for (int i = threadIdx.x; i < n; i += 256)
    prow[i] = (bf16_t)(__expf(row[i] - m) * inv);
}

__global__ __launch_bounds__(256) void add_layernorm(
    const float* __restrict__ x, const float* __restrict__ y,
    const float* __restrict__ g, const float* __restrict__ b,
    float* __restrict__ outF, bf16_t* __restrict__ outB) {
  __shared__ float red[8];
  const long long base = (long long)blockIdx.x * 512;
  const int t = threadIdx.x;
  const float a0 = x[base + t] + y[base + t];
  const float a1 = x[base + t + 256] + y[base + t + 256];
  const float s  = block_sum(a0 + a1, red);
  const float ss = block_sum(a0 * a0 + a1 * a1, red);
  const float mean = s * (1.f / 512.f);
  const float var  = ss * (1.f / 512.f) - mean * mean;
  const float rstd = rsqrtf(var + 1e-5f);
  const float o0 = (a0 - mean) * rstd * g[t] + b[t];
  const float o1 = (a1 - mean) * rstd * g[t + 256] + b[t + 256];
  outF[base + t]       = o0;
  outF[base + t + 256] = o1;
  if (outB) { outB[base + t] = (bf16_t)o0; outB[base + t + 256] = (bf16_t)o1; }
}

__global__ __launch_bounds__(256) void convert_bf16(const float* __restrict__ in,
                                                    bf16_t* __restrict__ out,
                                                    long long n) {
  for (long long i = (long long)blockIdx.x * 256 + threadIdx.x; i < n;
       i += (long long)gridDim.x * 256)
    out[i] = (bf16_t)in[i];
}

__global__ __launch_bounds__(256) void convtrans_f32_bf16(
    const float* __restrict__ in, int ild, bf16_t* __restrict__ out, int old_) {
  __shared__ float tile[32][33];
  const int c0 = blockIdx.x * 32, r0 = blockIdx.y * 32;
  for (int rr = threadIdx.y; rr < 32; rr += 8)
    tile[rr][threadIdx.x] = in[(long long)(r0 + rr) * ild + c0 + threadIdx.x];
  __syncthreads();
  for (int rr = threadIdx.y; rr < 32; rr += 8)
    out[(long long)(c0 + rr) * old_ + r0 + threadIdx.x] =
        (bf16_t)tile[threadIdx.x][rr];
}

__global__ __launch_bounds__(256) void transpose_bf16(
    const bf16_t* __restrict__ in, long long istride, int ild,
    bf16_t* __restrict__ out, long long ostride, int old_) {
  __shared__ bf16_t tile[32][33];
  const int c0 = blockIdx.x * 32, r0 = blockIdx.y * 32;
  in  += (long long)blockIdx.z * istride;
  out += (long long)blockIdx.z * ostride;
  for (int rr = threadIdx.y; rr < 32; rr += 8)
    tile[rr][threadIdx.x] = in[(long long)(r0 + rr) * ild + c0 + threadIdx.x];
  __syncthreads();
  for (int rr = threadIdx.y; rr < 32; rr += 8)
    out[(long long)(c0 + rr) * old_ + r0 + threadIdx.x] = tile[threadIdx.x][rr];
}

extern "C" void kernel_launch(void* const* d_in, const int* in_sizes, int n_in,
                              void* d_out, int out_size, void* d_ws, size_t ws_size,
                              hipStream_t stream) {
  (void)in_sizes; (void)n_in; (void)out_size; (void)ws_size;
  const float* x    = (const float*)d_in[0];
  const float* geo  = (const float*)d_in[1];
  const float* wqkv = (const float*)d_in[2];
  const float* bqkv = (const float*)d_in[3];
  const float* wout = (const float*)d_in[4];
  const float* bout = (const float*)d_in[5];
  const float* w1   = (const float*)d_in[6];
  const float* b1   = (const float*)d_in[7];
  const float* w2   = (const float*)d_in[8];
  const float* b2   = (const float*)d_in[9];
  const float* ln1g = (const float*)d_in[10];
  const float* ln1b = (const float*)d_in[11];
  const float* ln2g = (const float*)d_in[12];
  const float* ln2b = (const float*)d_in[13];

  const long long M = 16384;  // B*S
  const int S = 2048, Bt = 8, H = 512;

  char* ws = (char*)d_ws;
  size_t off = 0;
  auto take = [&](size_t bytes) {
    size_t o = off; off += (bytes + 255) & ~(size_t)255; return o;
  };
  bf16_t* xb    = (bf16_t*)(ws + take((size_t)M * H * 2));
  bf16_t* wqkvT = (bf16_t*)(ws + take((size_t)1536 * 512 * 2));
  bf16_t* woutT = (bf16_t*)(ws + take((size_t)512 * 512 * 2));
  bf16_t* w1T   = (bf16_t*)(ws + take((size_t)2048 * 512 * 2));
  bf16_t* w2T   = (bf16_t*)(ws + take((size_t)512 * 2048 * 2));
  bf16_t* qkvb  = (bf16_t*)(ws + take((size_t)M * 1536 * 2));
  bf16_t* vt    = (bf16_t*)(ws + take((size_t)Bt * 512 * 2048 * 2));
  float*  Smat  = (float*) (ws + take((size_t)Bt * S * S * 4));
  bf16_t* Pmat  = (bf16_t*)(ws + take((size_t)Bt * S * S * 2));
  bf16_t* aob   = (bf16_t*)(ws + take((size_t)M * H * 2));
  float*  proj  = (float*) (ws + take((size_t)M * H * 4));
  float*  x1    = (float*) (ws + take((size_t)M * H * 4));
  bf16_t* x1b   = (bf16_t*)(ws + take((size_t)M * H * 2));
  bf16_t* hbuf  = (bf16_t*)(ws + take((size_t)M * 2048 * 2));
  float*  fbuf  = (float*) (ws + take((size_t)M * H * 4));

  const dim3 blk256(256), tblk(32, 8);
  const float scale = 1.0f / sqrtf((float)H);

  convert_bf16<<<8192, blk256, 0, stream>>>(x, xb, M * H);
  convtrans_f32_bf16<<<dim3(48, 16), tblk, 0, stream>>>(wqkv, 1536, wqkvT, 512);
  convtrans_f32_bf16<<<dim3(16, 16), tblk, 0, stream>>>(wout, 512, woutT, 512);
  convtrans_f32_bf16<<<dim3(64, 16), tblk, 0, stream>>>(w1, 2048, w1T, 512);
  convtrans_f32_bf16<<<dim3(16, 64), tblk, 0, stream>>>(w2, 512, w2T, 2048);

  gemm_bf16_wmma<EPI_BIAS><<<dim3(12, 256, 1), blk256, 0, stream>>>(
      xb, 0, 512, wqkvT, 0, 512, bqkv, nullptr, qkvb, 0, 1536, 512,
      nullptr, 0, 0.f);

  transpose_bf16<<<dim3(16, 64, 8), tblk, 0, stream>>>(
      qkvb + 1024, (long long)S * 1536, 1536, vt, (long long)512 * 2048, 2048);

  gemm_bf16_wmma<EPI_ATTN><<<dim3(16, 32, 8), blk256, 0, stream>>>(
      qkvb, (long long)S * 1536, 1536, qkvb + 512, (long long)S * 1536, 1536,
      nullptr, Smat, nullptr, (long long)S * S, S, 512,
      geo, (long long)S * 3, scale);

  softmax_rows<<<Bt * S, blk256, 0, stream>>>(Smat, Pmat, S);

  gemm_bf16_wmma<EPI_BIAS><<<dim3(4, 32, 8), blk256, 0, stream>>>(
      Pmat, (long long)S * S, S, vt, (long long)512 * 2048, 2048, nullptr,
      nullptr, aob, (long long)S * 512, 512, 2048, nullptr, 0, 0.f);

  gemm_bf16_wmma<EPI_BIAS><<<dim3(4, 256, 1), blk256, 0, stream>>>(
      aob, 0, 512, woutT, 0, 512, bout, proj, nullptr, 0, 512, 512,
      nullptr, 0, 0.f);

  add_layernorm<<<(int)M, blk256, 0, stream>>>(x, proj, ln1g, ln1b, x1, x1b);

  gemm_bf16_wmma<EPI_RELU><<<dim3(16, 256, 1), blk256, 0, stream>>>(
      x1b, 0, 512, w1T, 0, 512, b1, nullptr, hbuf, 0, 2048, 512,
      nullptr, 0, 0.f);

  gemm_bf16_wmma<EPI_RELU><<<dim3(4, 256, 1), blk256, 0, stream>>>(
      hbuf, 0, 2048, w2T, 0, 2048, b2, fbuf, nullptr, 0, 512, 2048,
      nullptr, 0, 0.f);

  add_layernorm<<<(int)M, blk256, 0, stream>>>(x1, fbuf, ln2g, ln2b,
                                               (float*)d_out, nullptr);
}
